// MultiHeadAttentionSlow_18494129176736
// MI455X (gfx1250) — matrix operand order repaired and checked by reading, not compiled, not run on hardware
//
#include <hip/hip_runtime.h>

// ---------------------------------------------------------------------------
// MHA forward for MI455X (gfx1250, wave32, WMMA bf16 path).
//   B=4, S=2048, D=1024, H=16, h_dim=64
// Pipeline: qkv_proj (bf16 WMMA GEMM) -> flash attention (WMMA + online
// softmax, async K-tile staging) -> out_proj (bf16 WMMA GEMM, fp32 out).
// ---------------------------------------------------------------------------

#define D_MODEL 1024
#define SEQ     2048
#define NB      4
#define NHEAD   16
#define HDIM    64
#define LDSP    72   // padded LDS row stride in bf16 elems (144B, 16B aligned)

typedef __attribute__((ext_vector_type(16))) __bf16 v16bf;
typedef __attribute__((ext_vector_type(8)))  __bf16 v8bf;
typedef __attribute__((ext_vector_type(8)))  float  v8f;

union Frag {
    v16bf v;
    v8bf  h[2];
};

static __device__ __forceinline__ __bf16 f2bf(float f) {
    unsigned u = __builtin_bit_cast(unsigned, f);
    u += 0x7FFFu + ((u >> 16) & 1u);          // round-to-nearest-even
    return __builtin_bit_cast(__bf16, (unsigned short)(u >> 16));
}

// pack two fp32 -> two bf16 in one dword via v_perm_b32
static __device__ __forceinline__ unsigned pk2bf(float a, float b) {
    unsigned ua = __builtin_bit_cast(unsigned, a);
    unsigned ub = __builtin_bit_cast(unsigned, b);
    ua += 0x7FFFu + ((ua >> 16) & 1u);
    ub += 0x7FFFu + ((ub >> 16) & 1u);
    // result bytes: [0,1] = ua[2,3], [2,3] = ub[2,3]
    return __builtin_amdgcn_perm(ub, ua, 0x07060302u);
}

static __device__ __forceinline__ v8f zero8() {
    v8f z = {0.f, 0.f, 0.f, 0.f, 0.f, 0.f, 0.f, 0.f};
    return z;
}

// CDNA5 async global->LDS copy (16B per lane), tracked by ASYNCcnt.
// lds_off = low 32 bits of flat shared address (ISA 10.2: LDS aperture uses addr[31:0]).
static __device__ __forceinline__ void async_copy_b128(void* lds_ptr, const void* gptr) {
    unsigned lds_off = (unsigned)(unsigned long long)lds_ptr;
    asm volatile("global_load_async_to_lds_b128 %0, %1, off"
                 :: "v"(lds_off), "v"(gptr) : "memory");
}
static __device__ __forceinline__ void async_wait0() {
    asm volatile("s_wait_asynccnt 0x0" ::: "memory");
}

#define WMMA_BF16(A, B, C) \
    __builtin_amdgcn_wmma_f32_16x16x32_bf16(false, (A), false, (B), (short)0, (C), false, false)

// ---------------------------------------------------------------------------
// Kernel 1: per-head QKV projection.
//   X[8192,1024] (fp32) x W[h][1024,64] (fp32) + b  ->  bf16 [B,H,S,64] in ws.
// blockIdx.z selects query/key/value. Workgroup tile 128(M) x 128(N),
// 8 waves (4x2), each wave does 2x4 WMMA subtiles, K staged 64 at a time.
// ---------------------------------------------------------------------------
__global__ __launch_bounds__(256) void qkv_proj_kernel(
    const float* __restrict__ q_in, const float* __restrict__ k_in,
    const float* __restrict__ v_in,
    const float* __restrict__ Wq, const float* __restrict__ bq,
    const float* __restrict__ Wk, const float* __restrict__ bk,
    const float* __restrict__ Wv, const float* __restrict__ bv,
    __bf16* __restrict__ qkv_ws)
{
    const int which = blockIdx.z;
    const float* X    = (which == 0) ? q_in : (which == 1) ? k_in : v_in;
    const float* W    = (which == 0) ? Wq   : (which == 1) ? Wk   : Wv;
    const float* bias = (which == 0) ? bq   : (which == 1) ? bk   : bv;
    __bf16* O = qkv_ws + (size_t)which * ((size_t)NB * NHEAD * SEQ * HDIM);

    __shared__ __align__(16) __bf16 Atile[128][LDSP];   // [M][K]
    __shared__ __align__(16) __bf16 Btile[128][LDSP];   // [N][K] (transposed)

    const int tid  = threadIdx.x;
    const int lane = tid & 31;
    const int wave = tid >> 5;
    const int wm   = wave & 3;   // wave M position (0..3)
    const int wn   = wave >> 2;  // wave N position (0..1)
    const int Mbase = blockIdx.x * 128;
    const int Nbase = blockIdx.y * 128;
    const int halfsel = lane >> 4;

    v8f acc[2][4];
    for (int mi = 0; mi < 2; ++mi)
        for (int ni = 0; ni < 4; ++ni) acc[mi][ni] = zero8();

    for (int k0 = 0; k0 < D_MODEL; k0 += 64) {
        __syncthreads();
        // --- stage A: 128 rows x 64 K of fp32 input -> bf16 LDS -------------
        for (int it = 0; it < 8; ++it) {
            int idx = tid + it * 256;           // 0..2047 over (row, col4)
            int row = idx >> 4;
            int c4  = (idx & 15) << 2;
            const float4 f =
                *(const float4*)&X[(size_t)(Mbase + row) * D_MODEL + k0 + c4];
            uint2 p;
            p.x = pk2bf(f.x, f.y);
            p.y = pk2bf(f.z, f.w);
            *(uint2*)&Atile[row][c4] = p;
        }
        // --- stage B transposed: Btile[n][k] = W[h][k0+k][n&63] -------------
        for (int it = 0; it < 8; ++it) {
            int idx = tid + it * 256;           // 0..2047 over (k, n4)
            int k = idx >> 5;                   // 0..63
            int n = (idx & 31) << 2;            // 0..124
            int col = Nbase + n;
            int h = col >> 6;
            const float4 f = *(const float4*)&W[((size_t)h * D_MODEL + (k0 + k)) * HDIM + (col & 63)];
            Btile[n + 0][k] = f2bf(f.x);
            Btile[n + 1][k] = f2bf(f.y);
            Btile[n + 2][k] = f2bf(f.z);
            Btile[n + 3][k] = f2bf(f.w);
        }
        __syncthreads();

        // --- 2 K-chunks of 32, 8 WMMAs each ---------------------------------
        for (int kc = 0; kc < 64; kc += 32) {
            Frag a[2], b[4];
            const int arow0 = wm * 32 + (lane & 15);
            const int akoff = kc + ((lane < 16) ? 0 : 8);
            for (int mi = 0; mi < 2; ++mi) {
                int r = arow0 + mi * 16;
                a[mi].h[0] = *(const v8bf*)&Atile[r][akoff];
                a[mi].h[1] = *(const v8bf*)&Atile[r][akoff + 16];
            }
            const int bkoff = kc + ((lane < 16) ? 0 : 8);
            for (int ni = 0; ni < 4; ++ni) {
                int c = wn * 64 + ni * 16 + (lane & 15);
                b[ni].h[0] = *(const v8bf*)&Btile[c][bkoff];
                b[ni].h[1] = *(const v8bf*)&Btile[c][bkoff + 16];
            }
            for (int mi = 0; mi < 2; ++mi)
                for (int ni = 0; ni < 4; ++ni)
                    acc[mi][ni] = WMMA_BF16(a[mi].v, b[ni].v, acc[mi][ni]);
        }
    }

    // --- epilogue: +bias, convert, scatter into [B,H,S,64] ------------------
    for (int mi = 0; mi < 2; ++mi) {
        for (int ni = 0; ni < 4; ++ni) {
            int col = Nbase + wn * 64 + ni * 16 + (lane & 15);
            int h = col >> 6, kk = col & 63;
            float bb = bias[col];
            for (int i = 0; i < 8; ++i) {
                int row = Mbase + wm * 32 + mi * 16 + i + halfsel * 8;
                int bI = row >> 11, s = row & (SEQ - 1);
                O[(((size_t)(bI * NHEAD + h)) * SEQ + s) * HDIM + kk] =
                    f2bf(acc[mi][ni][i] + bb);
            }
        }
    }
}

// ---------------------------------------------------------------------------
// Kernel 2: flash attention per (b,h). Wave owns 16 query rows, streams over
// 64-key blocks staged in LDS. K tile staged with async global->LDS copies;
// V transposed through VGPRs. Online softmax; scores never touch memory.
// ---------------------------------------------------------------------------
__global__ __launch_bounds__(256) void attn_kernel(
    const __bf16* __restrict__ Q, const __bf16* __restrict__ K,
    const __bf16* __restrict__ V, __bf16* __restrict__ Aout)
{
    __shared__ __align__(16) __bf16 Ktile[64][LDSP];       // [key][feat]
    __shared__ __align__(16) __bf16 Vtile[64][LDSP];       // [feat][key] (transposed)
    __shared__ __align__(16) __bf16 Ptile[8][16][LDSP];    // per-wave P round-trip

    const int tid  = threadIdx.x;
    const int lane = tid & 31;
    const int wave = tid >> 5;
    const int halfsel = lane >> 4;
    const int bh = blockIdx.y;                 // 0..63
    const int b = bh >> 4, h = bh & 15;
    const int rowBase = blockIdx.x * 128 + wave * 16;
    const size_t headOff = (size_t)bh * SEQ * HDIM;

    // Q fragments held in registers for the whole key loop.
    Frag aq[2];
    {
        int qrow = rowBase + (lane & 15);
        const __bf16* qp = Q + headOff + (size_t)qrow * HDIM;
        for (int kc2 = 0; kc2 < 2; ++kc2) {
            int koff = kc2 * 32 + ((lane < 16) ? 0 : 8);
            aq[kc2].h[0] = *(const v8bf*)&qp[koff];
            aq[kc2].h[1] = *(const v8bf*)&qp[koff + 16];
        }
    }

    v8f o[4];
    for (int ni = 0; ni < 4; ++ni) o[ni] = zero8();
    float mrow[8], lrow[8];
    for (int i = 0; i < 8; ++i) { mrow[i] = -1e30f; lrow[i] = 0.f; }

    const float scale = 0.125f;   // 1/sqrt(64)

    for (int kb = 0; kb < SEQ; kb += 64) {
        __syncthreads();
        // --- stage K via async global->LDS (b128/lane), V via VGPR transpose -
        for (int it = 0; it < 2; ++it) {
            int idx = tid + it * 256;           // 0..511 over (key, featgrp)
            int key = idx >> 3;                 // 0..63
            int g = (idx & 7) << 3;             // 0,8,..,56
            async_copy_b128(&Ktile[key][g],
                            &K[headOff + (size_t)(kb + key) * HDIM + g]);
            v8bf vv = *(const v8bf*)&V[headOff + (size_t)(kb + key) * HDIM + g];
            for (int j = 0; j < 8; ++j) Vtile[g + j][key] = vv[j];
        }
        async_wait0();
        __syncthreads();

        // --- scores: 16x64 tile = Q(16x64) * K^T, 8 WMMAs -------------------
        v8f sc[4];
        for (int ni = 0; ni < 4; ++ni) sc[ni] = zero8();
        for (int kc2 = 0; kc2 < 2; ++kc2) {
            int bkoff = kc2 * 32 + ((lane < 16) ? 0 : 8);
            for (int ni = 0; ni < 4; ++ni) {
                Frag bk;
                int c = ni * 16 + (lane & 15);
                bk.h[0] = *(const v8bf*)&Ktile[c][bkoff];
                bk.h[1] = *(const v8bf*)&Ktile[c][bkoff + 16];
                sc[ni] = WMMA_BF16(aq[kc2].v, bk.v, sc[ni]);
            }
        }

        // --- online softmax per row slot (16-lane half reductions) ----------
        for (int i = 0; i < 8; ++i) {
            float v0 = sc[0][i] * scale, v1 = sc[1][i] * scale;
            float v2 = sc[2][i] * scale, v3 = sc[3][i] * scale;
            float mx = fmaxf(fmaxf(v0, v1), fmaxf(v2, v3));
            for (int d = 1; d < 16; d <<= 1)
                mx = fmaxf(mx, __shfl_xor(mx, d, 32));
            float mnew = fmaxf(mrow[i], mx);
            float corr = __expf(mrow[i] - mnew);
            float p0 = __expf(v0 - mnew), p1 = __expf(v1 - mnew);
            float p2 = __expf(v2 - mnew), p3 = __expf(v3 - mnew);
            float rs = p0 + p1 + p2 + p3;
            for (int d = 1; d < 16; d <<= 1)
                rs += __shfl_xor(rs, d, 32);
            lrow[i] = lrow[i] * corr + rs;
            mrow[i] = mnew;
            for (int ni = 0; ni < 4; ++ni) o[ni][i] *= corr;
            int prow = i + halfsel * 8;
            unsigned pk01 = pk2bf(p0, p1);      // not adjacent cols; store halves
            Ptile[wave][prow][ 0 + (lane & 15)] = __builtin_bit_cast(__bf16, (unsigned short)(pk01 & 0xFFFFu));
            Ptile[wave][prow][16 + (lane & 15)] = __builtin_bit_cast(__bf16, (unsigned short)(pk01 >> 16));
            unsigned pk23 = pk2bf(p2, p3);
            Ptile[wave][prow][32 + (lane & 15)] = __builtin_bit_cast(__bf16, (unsigned short)(pk23 & 0xFFFFu));
            Ptile[wave][prow][48 + (lane & 15)] = __builtin_bit_cast(__bf16, (unsigned short)(pk23 >> 16));
        }

        // --- PV: out += P(16x64) * V(64x64), 8 WMMAs ------------------------
        for (int kc2 = 0; kc2 < 2; ++kc2) {
            Frag ap;
            int prow = lane & 15;
            int akoff = kc2 * 32 + ((lane < 16) ? 0 : 8);
            ap.h[0] = *(const v8bf*)&Ptile[wave][prow][akoff];
            ap.h[1] = *(const v8bf*)&Ptile[wave][prow][akoff + 16];
            int bkoff = kc2 * 32 + ((lane < 16) ? 0 : 8);
            for (int ni = 0; ni < 4; ++ni) {
                Frag bv_;
                int c = ni * 16 + (lane & 15);
                bv_.h[0] = *(const v8bf*)&Vtile[c][bkoff];
                bv_.h[1] = *(const v8bf*)&Vtile[c][bkoff + 16];
                o[ni] = WMMA_BF16(ap.v, bv_.v, o[ni]);
            }
        }
    }

    // --- normalize and write concat layout [B,S,H*64] -----------------------
    for (int ni = 0; ni < 4; ++ni) {
        int feat = ni * 16 + (lane & 15);
        for (int i = 0; i < 8; ++i) {
            int srow = rowBase + i + halfsel * 8;
            float val = o[ni][i] / lrow[i];
            Aout[((size_t)(b * SEQ + srow)) * D_MODEL + h * HDIM + feat] = f2bf(val);
        }
    }
}

// ---------------------------------------------------------------------------
// Kernel 3: output projection. bf16 A [8192,1024] x Wp(fp32->bf16) + bp,
// fp32 output. A tile staged with async global->LDS copies.
// ---------------------------------------------------------------------------
__global__ __launch_bounds__(256) void out_proj_kernel(
    const __bf16* __restrict__ A, const float* __restrict__ Wp,
    const float* __restrict__ bp, float* __restrict__ out)
{
    __shared__ __align__(16) __bf16 Atile[128][LDSP];
    __shared__ __align__(16) __bf16 Btile[128][LDSP];

    const int tid  = threadIdx.x;
    const int lane = tid & 31;
    const int wave = tid >> 5;
    const int wm   = wave & 3;
    const int wn   = wave >> 2;
    const int Mbase = blockIdx.x * 128;
    const int Nbase = blockIdx.y * 128;
    const int halfsel = lane >> 4;

    v8f acc[2][4];
    for (int mi = 0; mi < 2; ++mi)
        for (int ni = 0; ni < 4; ++ni) acc[mi][ni] = zero8();

    for (int k0 = 0; k0 < D_MODEL; k0 += 64) {
        __syncthreads();
        // A already bf16: async global->LDS, 16B per lane per iteration
        for (int it = 0; it < 4; ++it) {
            int idx = tid + it * 256;           // 0..1023 over (row, featgrp)
            int row = idx >> 3;
            int g = (idx & 7) << 3;
            async_copy_b128(&Atile[row][g],
                            &A[(size_t)(Mbase + row) * D_MODEL + k0 + g]);
        }
        // B transposed: Btile[n][k] = Wp[(k0+k)*D + Nbase+n] (fp32 -> bf16)
        for (int it = 0; it < 8; ++it) {
            int idx = tid + it * 256;
            int k = idx >> 5;
            int n = (idx & 31) << 2;
            const float4 f =
                *(const float4*)&Wp[(size_t)(k0 + k) * D_MODEL + Nbase + n];
            Btile[n + 0][k] = f2bf(f.x);
            Btile[n + 1][k] = f2bf(f.y);
            Btile[n + 2][k] = f2bf(f.z);
            Btile[n + 3][k] = f2bf(f.w);
        }
        async_wait0();
        __syncthreads();

        for (int kc = 0; kc < 64; kc += 32) {
            Frag a[2], b[4];
            const int arow0 = wm * 32 + (lane & 15);
            const int akoff = kc + ((lane < 16) ? 0 : 8);
            for (int mi = 0; mi < 2; ++mi) {
                int r = arow0 + mi * 16;
                a[mi].h[0] = *(const v8bf*)&Atile[r][akoff];
                a[mi].h[1] = *(const v8bf*)&Atile[r][akoff + 16];
            }
            const int bkoff = kc + ((lane < 16) ? 0 : 8);
            for (int ni = 0; ni < 4; ++ni) {
                int c = wn * 64 + ni * 16 + (lane & 15);
                b[ni].h[0] = *(const v8bf*)&Btile[c][bkoff];
                b[ni].h[1] = *(const v8bf*)&Btile[c][bkoff + 16];
            }
            for (int mi = 0; mi < 2; ++mi)
                for (int ni = 0; ni < 4; ++ni)
                    acc[mi][ni] = WMMA_BF16(a[mi].v, b[ni].v, acc[mi][ni]);
        }
    }

    for (int mi = 0; mi < 2; ++mi) {
        for (int ni = 0; ni < 4; ++ni) {
            int col = Nbase + wn * 64 + ni * 16 + (lane & 15);
            float bb = bp[col];
            for (int i = 0; i < 8; ++i) {
                int row = Mbase + wm * 32 + mi * 16 + i + halfsel * 8;
                out[(size_t)row * D_MODEL + col] = acc[mi][ni][i] + bb;
            }
        }
    }
}

// ---------------------------------------------------------------------------
extern "C" void kernel_launch(void* const* d_in, const int* in_sizes, int n_in,
                              void* d_out, int out_size, void* d_ws, size_t ws_size,
                              hipStream_t stream)
{
    (void)in_sizes; (void)n_in; (void)out_size; (void)ws_size;

    const float* query = (const float*)d_in[0];
    const float* key_  = (const float*)d_in[1];
    const float* value = (const float*)d_in[2];
    const float* Wq = (const float*)d_in[3];
    const float* bq = (const float*)d_in[4];
    const float* Wk = (const float*)d_in[5];
    const float* bk = (const float*)d_in[6];
    const float* Wv = (const float*)d_in[7];
    const float* bv = (const float*)d_in[8];
    const float* Wp = (const float*)d_in[9];
    const float* bp = (const float*)d_in[10];
    float* out = (float*)d_out;

    __bf16* ws = (__bf16*)d_ws;
    const size_t HE = (size_t)NB * NHEAD * SEQ * HDIM;   // 8,388,608 elems
    __bf16* qw = ws;              // [B,H,S,64] bf16
    __bf16* kw = ws + HE;
    __bf16* vw = ws + 2 * HE;
    __bf16* aw = ws + 3 * HE;     // concat attention out [B,S,1024] bf16

    // 1) QKV projections (z = 0:q, 1:k, 2:v)
    qkv_proj_kernel<<<dim3(64, 8, 3), 256, 0, stream>>>(
        query, key_, value, Wq, bq, Wk, bk, Wv, bv, qw);

    // 2) flash attention over 64 (b,h) pairs, 128 query rows per workgroup
    attn_kernel<<<dim3(SEQ / 128, NB * NHEAD), 256, 0, stream>>>(qw, kw, vw, aw);

    // 3) output projection
    out_proj_kernel<<<dim3(64, 8), 256, 0, stream>>>(aw, Wp, bp, out);
}
